// FeedForward_19413252178269
// MI455X (gfx1250) — matrix operand order repaired and checked by reading, not compiled, not run on hardware
//
#include <hip/hip_runtime.h>

// ---------------------------------------------------------------------------
// Types
// ---------------------------------------------------------------------------
typedef __bf16 bf16_t;
typedef __attribute__((ext_vector_type(4)))  bf16_t v4bf;
typedef __attribute__((ext_vector_type(8)))  bf16_t v8bf;
typedef __attribute__((ext_vector_type(16))) bf16_t v16bf;
typedef __attribute__((ext_vector_type(8)))  float  v8f;

#define D_DIM 2048
#define F_DIM 8192
#define R_DIM 16
#define SCALING 2.0f
#define GAMMA 0.1f

#define BM 128
#define BN 128
#define WM 64
#define WN 32
#define KC 32

// ---------------------------------------------------------------------------
// Helpers
// ---------------------------------------------------------------------------
__device__ __forceinline__ v8f wmma_bf16(v16bf a, v16bf b, v8f c) {
  // v_wmma_f32_16x16x32_bf16  (neg_a, A, neg_b, B, c_mod, C, reuse_a, reuse_b)
  return __builtin_amdgcn_wmma_f32_16x16x32_bf16(false, a, false, b,
                                                 (short)0, c, false, false);
}

union FragU { v16bf v; v8bf h[2]; };

// A fragment, 16x32 bf16 (MxK), row-major source, ld = row stride (elements).
// lane<16 : row=lane,    K in {kk..kk+7, kk+16..kk+23}
// lane>=16: row=lane-16, K in {kk+8..kk+15, kk+24..kk+31}
__device__ __forceinline__ v16bf load_a_frag(const bf16_t* __restrict__ base,
                                             int ld, int row, int kk, int hi) {
  const bf16_t* p = base + (size_t)row * ld + kk + hi * 8;
  FragU u;
  u.h[0] = *reinterpret_cast<const v8bf*>(p);
  u.h[1] = *reinterpret_cast<const v8bf*>(p + 16);
  return u.v;
}

// B fragment, 32x16 bf16 (KxN), from transposed weights WT[N][K] row-major.
// lane<16 : col=lane,    K in {kk..kk+7, kk+16..kk+23}
// lane>=16: col=lane-16, K in {kk+8..kk+15, kk+24..kk+31}
__device__ __forceinline__ v16bf load_b_frag(const bf16_t* __restrict__ wt,
                                             int ld, int col, int kk, int hi) {
  const bf16_t* p = wt + (size_t)col * ld + kk + hi * 8;
  FragU u;
  u.h[0] = *reinterpret_cast<const v8bf*>(p);
  u.h[1] = *reinterpret_cast<const v8bf*>(p + 16);
  return u.v;
}

// ERA(x) = gelu_tanh(x) + GAMMA * softplus(x)
__device__ __forceinline__ float era_f(float x) {
  const float k0 = 0.7978845608028654f; // sqrt(2/pi)
  float x3 = x * x * x;
  float t = tanhf(k0 * (x + 0.044715f * x3));
  float gelu = 0.5f * x * (1.0f + t);
  float sp = (x > 20.0f) ? x : log1pf(expf(x));
  return gelu + GAMMA * sp;
}

// ---------------------------------------------------------------------------
// Conversion kernels
// ---------------------------------------------------------------------------
__global__ __launch_bounds__(256) void cvt_f32_to_bf16(
    const float* __restrict__ src, bf16_t* __restrict__ dst, size_t n) {
  size_t i = ((size_t)blockIdx.x * blockDim.x + threadIdx.x) * 4;
  if (i + 3 < n) {
    float4 v = *reinterpret_cast<const float4*>(src + i);
    v4bf o;
    o[0] = (bf16_t)v.x; o[1] = (bf16_t)v.y;
    o[2] = (bf16_t)v.z; o[3] = (bf16_t)v.w;
    *reinterpret_cast<v4bf*>(dst + i) = o;
  }
}

// src[R][C] fp32 -> dst[C][R] bf16 (tiled transpose through LDS)
__global__ __launch_bounds__(256) void transpose_f32_to_bf16(
    const float* __restrict__ src, bf16_t* __restrict__ dst, int R, int C) {
  __shared__ bf16_t tile[32][33];
  int c0 = blockIdx.x * 32;
  int r0 = blockIdx.y * 32;
  for (int i = threadIdx.y; i < 32; i += 8)
    tile[i][threadIdx.x] =
        (bf16_t)src[(size_t)(r0 + i) * C + (c0 + threadIdx.x)];
  __syncthreads();
  for (int i = threadIdx.y; i < 32; i += 8)
    dst[(size_t)(c0 + i) * R + (r0 + threadIdx.x)] = tile[threadIdx.x][i];
}

// ---------------------------------------------------------------------------
// Rank-16 LoRA down projection: t = x @ Wdn + bdn  (f32, tiny)
// ---------------------------------------------------------------------------
__global__ __launch_bounds__(256) void lora_down(
    const float* __restrict__ x,
    const float* __restrict__ Wgd, const float* __restrict__ bgd,
    const float* __restrict__ Wud, const float* __restrict__ bud,
    float* __restrict__ tg, float* __restrict__ tu, int M) {
  int tid = blockIdx.x * blockDim.x + threadIdx.x;
  int r = tid & 15;
  int m = tid >> 4;
  if (m >= M) return;
  const float* xp = x + (size_t)m * D_DIM;
  float ag = 0.f, au = 0.f;
  for (int d = 0; d < D_DIM; ++d) {
    float xv = xp[d];
    ag = fmaf(xv, Wgd[d * R_DIM + r], ag);
    au = fmaf(xv, Wud[d * R_DIM + r], au);
  }
  tg[m * R_DIM + r] = ag + bgd[r];
  tu[m * R_DIM + r] = au + bud[r];
}

// ---------------------------------------------------------------------------
// Fused gate/up GEMM + LoRA epilogue + ERA + elementwise mul  -> h (bf16)
//   gate_pre = x@Wg + bg + 2*(tg@Wgu + bgu) ; up_pre likewise
//   h = ERA(gate_pre) * up_pre
// ---------------------------------------------------------------------------
__global__ __launch_bounds__(256) void fused_gate_up(
    const bf16_t* __restrict__ xb,
    const bf16_t* __restrict__ WgT, const bf16_t* __restrict__ WuT,
    const float* __restrict__ bg, const float* __restrict__ bgu,
    const float* __restrict__ bu, const float* __restrict__ buu,
    const float* __restrict__ tg, const float* __restrict__ tu,
    const float* __restrict__ Wgu, const float* __restrict__ Wuu,
    bf16_t* __restrict__ h) {
  int lane = threadIdx.x & 31;
  int wave = threadIdx.x >> 5;
  int waveRow = wave >> 2;        // 0..1
  int waveCol = wave & 3;         // 0..3
  int lane15 = lane & 15;
  int hi = lane >> 4;
  int rowBase = blockIdx.y * BM + waveRow * WM;
  int colBase = blockIdx.x * BN + waveCol * WN;

  v8f accg[4][2], accu[4][2];
#pragma unroll
  for (int i = 0; i < 4; ++i)
#pragma unroll
    for (int j = 0; j < 2; ++j) { accg[i][j] = (v8f)0.0f; accu[i][j] = (v8f)0.0f; }

  for (int kk = 0; kk < D_DIM; kk += KC) {
    v16bf bgf[2], buf2[2];
#pragma unroll
    for (int j = 0; j < 2; ++j) {
      int col = colBase + j * 16 + lane15;
      bgf[j]  = load_b_frag(WgT, D_DIM, col, kk, hi);
      buf2[j] = load_b_frag(WuT, D_DIM, col, kk, hi);
    }
#pragma unroll
    for (int i = 0; i < 4; ++i) {
      v16bf af = load_a_frag(xb, D_DIM, rowBase + i * 16 + lane15, kk, hi);
#pragma unroll
      for (int j = 0; j < 2; ++j) {
        accg[i][j] = wmma_bf16(af, bgf[j],  accg[i][j]);
        accu[i][j] = wmma_bf16(af, buf2[j], accu[i][j]);
      }
    }
  }

  // Epilogue: LoRA rank-16 correction + biases + ERA + mul, store bf16 h.
#pragma unroll
  for (int j = 0; j < 2; ++j) {
    int fcol = colBase + j * 16 + lane15;
    float bgv  = bg[fcol],  bguv = bgu[fcol];
    float buv  = bu[fcol],  buuv = buu[fcol];
    float wgR[R_DIM], wuR[R_DIM];
#pragma unroll
    for (int r = 0; r < R_DIM; ++r) {
      wgR[r] = Wgu[(size_t)r * F_DIM + fcol];
      wuR[r] = Wuu[(size_t)r * F_DIM + fcol];
    }
#pragma unroll
    for (int i = 0; i < 4; ++i) {
#pragma unroll
      for (int e = 0; e < 8; ++e) {
        int row = rowBase + i * 16 + e + hi * 8;
        const float* tgp = tg + (size_t)row * R_DIM;
        const float* tup = tu + (size_t)row * R_DIM;
        float lg = 0.f, lu = 0.f;
#pragma unroll
        for (int r = 0; r < R_DIM; ++r) {
          lg = fmaf(tgp[r], wgR[r], lg);
          lu = fmaf(tup[r], wuR[r], lu);
        }
        float gp  = accg[i][j][e] + bgv + SCALING * (lg + bguv);
        float upv = accu[i][j][e] + buv + SCALING * (lu + buuv);
        h[(size_t)row * F_DIM + fcol] = (bf16_t)(era_f(gp) * upv);
      }
    }
  }
}

// ---------------------------------------------------------------------------
// Down projection: out = h @ Wd + bd   (K = F = 8192, N = D = 2048)
// ---------------------------------------------------------------------------
__global__ __launch_bounds__(256) void down_proj(
    const bf16_t* __restrict__ h, const bf16_t* __restrict__ WdT,
    const float* __restrict__ bd, float* __restrict__ out) {
  int lane = threadIdx.x & 31;
  int wave = threadIdx.x >> 5;
  int waveRow = wave >> 2;
  int waveCol = wave & 3;
  int lane15 = lane & 15;
  int hi = lane >> 4;
  int rowBase = blockIdx.y * BM + waveRow * WM;
  int colBase = blockIdx.x * BN + waveCol * WN;

  v8f acc[4][2];
#pragma unroll
  for (int i = 0; i < 4; ++i)
#pragma unroll
    for (int j = 0; j < 2; ++j) acc[i][j] = (v8f)0.0f;

  for (int kk = 0; kk < F_DIM; kk += KC) {
    v16bf bf[2];
#pragma unroll
    for (int j = 0; j < 2; ++j)
      bf[j] = load_b_frag(WdT, F_DIM, colBase + j * 16 + lane15, kk, hi);
#pragma unroll
    for (int i = 0; i < 4; ++i) {
      v16bf af = load_a_frag(h, F_DIM, rowBase + i * 16 + lane15, kk, hi);
#pragma unroll
      for (int j = 0; j < 2; ++j)
        acc[i][j] = wmma_bf16(af, bf[j], acc[i][j]);
    }
  }

#pragma unroll
  for (int j = 0; j < 2; ++j) {
    int n = colBase + j * 16 + lane15;
    float bdv = bd[n];
#pragma unroll
    for (int i = 0; i < 4; ++i) {
#pragma unroll
      for (int e = 0; e < 8; ++e) {
        int row = rowBase + i * 16 + e + hi * 8;
        out[(size_t)row * D_DIM + n] = acc[i][j][e] + bdv;
      }
    }
  }
}

// ---------------------------------------------------------------------------
// Launch
// ---------------------------------------------------------------------------
extern "C" void kernel_launch(void* const* d_in, const int* in_sizes, int n_in,
                              void* d_out, int out_size, void* d_ws, size_t ws_size,
                              hipStream_t stream) {
  const float* x   = (const float*)d_in[0];
  const float* Wg  = (const float*)d_in[1];
  const float* bg  = (const float*)d_in[2];
  const float* Wgd = (const float*)d_in[3];
  const float* bgd = (const float*)d_in[4];
  const float* Wgu = (const float*)d_in[5];
  const float* bgu = (const float*)d_in[6];
  const float* Wu  = (const float*)d_in[7];
  const float* bu  = (const float*)d_in[8];
  const float* Wud = (const float*)d_in[9];
  const float* bud = (const float*)d_in[10];
  const float* Wuu = (const float*)d_in[11];
  const float* buu = (const float*)d_in[12];
  const float* Wd  = (const float*)d_in[13];
  const float* bd  = (const float*)d_in[14];
  float* out = (float*)d_out;

  const int M = in_sizes[0] / D_DIM;   // 8192 tokens

  // Workspace carve-up
  char* ws = (char*)d_ws;
  size_t o = 0;
  bf16_t* xb  = (bf16_t*)(ws + o); o += (size_t)M * D_DIM * sizeof(bf16_t);
  bf16_t* WgT = (bf16_t*)(ws + o); o += (size_t)F_DIM * D_DIM * sizeof(bf16_t);
  bf16_t* WuT = (bf16_t*)(ws + o); o += (size_t)F_DIM * D_DIM * sizeof(bf16_t);
  bf16_t* WdT = (bf16_t*)(ws + o); o += (size_t)D_DIM * F_DIM * sizeof(bf16_t);
  float*  tg  = (float*)(ws + o);  o += (size_t)M * R_DIM * sizeof(float);
  float*  tu  = (float*)(ws + o);  o += (size_t)M * R_DIM * sizeof(float);
  bf16_t* h   = (bf16_t*)(ws + o); o += (size_t)M * F_DIM * sizeof(bf16_t);
  (void)ws_size; (void)n_in; (void)out_size;

  // 1) x -> bf16
  {
    size_t n = (size_t)M * D_DIM;
    cvt_f32_to_bf16<<<(unsigned)(n / (256 * 4)), 256, 0, stream>>>(x, xb, n);
  }
  // 2) transpose+convert weights: W[K][N] -> WT[N][K] bf16
  transpose_f32_to_bf16<<<dim3(F_DIM / 32, D_DIM / 32), dim3(32, 8), 0, stream>>>(Wg, WgT, D_DIM, F_DIM);
  transpose_f32_to_bf16<<<dim3(F_DIM / 32, D_DIM / 32), dim3(32, 8), 0, stream>>>(Wu, WuT, D_DIM, F_DIM);
  transpose_f32_to_bf16<<<dim3(D_DIM / 32, F_DIM / 32), dim3(32, 8), 0, stream>>>(Wd, WdT, F_DIM, D_DIM);
  // 3) rank-16 LoRA down projections (f32)
  lora_down<<<(M * R_DIM) / 256, 256, 0, stream>>>(x, Wgd, bgd, Wud, bud, tg, tu, M);
  // 4) fused gate/up WMMA + epilogue -> h
  fused_gate_up<<<dim3(F_DIM / BN, M / BM), 256, 0, stream>>>(
      xb, WgT, WuT, bg, bgu, bu, buu, tg, tu, Wgu, Wuu, h);
  // 5) down projection WMMA -> out
  down_proj<<<dim3(D_DIM / BN, M / BM), 256, 0, stream>>>(h, WdT, bd, out);
}
